// VSSSBlock1D_25898652795557
// MI455X (gfx1250) — matrix operand order repaired and checked by reading, not compiled, not run on hardware
//
#include <hip/hip_runtime.h>
#include <cstddef>
#include <cstdint>

// ---------------------------------------------------------------------------
// Mamba-style SSM block for MI455X (gfx1250, wave32, WMMA).
// GEMMs: v_wmma_f32_16x16x32_bf16; bf16 tiles double-buffered in LDS; every
// fragment gather = 2x ds_load_b128; 4 WMMAs per wave per K-step.
// Scan: per-(b,d) recurrence; B/C tiles DMA'd with pipelined
// global_load_async_to_lds_b128 (ASYNCcnt double buffering).
// ---------------------------------------------------------------------------

#define BATCH   4
#define DMODEL  256
#define DSTATE  16
#define DINNER  512
#define DTRANK  16
#define SEQLEN  2048
#define XROWS   48           // DT_RANK + 2*D_STATE

typedef __attribute__((ext_vector_type(16))) __bf16 v16bf;
typedef __attribute__((ext_vector_type(8)))  __bf16 v8bf;
typedef __attribute__((ext_vector_type(2)))  __bf16 v2bf;
typedef __attribute__((ext_vector_type(8)))  float  v8f;

#define PITCH 40   // bf16 elems per LDS row: 80 B -> every 8-elem group 16B aligned

// workspace layout (float elements)
static constexpr size_t SZ_XZ = (size_t)BATCH * 2 * DINNER * SEQLEN;
static constexpr size_t SZ_U  = (size_t)BATCH * DINNER * SEQLEN;
static constexpr size_t SZ_XD = (size_t)BATCH * XROWS * SEQLEN;
static constexpr size_t OFF_XZ = 0;
static constexpr size_t OFF_U  = OFF_XZ + SZ_XZ;
static constexpr size_t OFF_XD = OFF_U  + SZ_U;
static constexpr size_t OFF_DL = OFF_XD + SZ_XD;
static constexpr size_t OFF_Y  = OFF_DL + SZ_U;

__device__ __forceinline__ float fast_sigmoid(float x) {
    return 1.0f / (1.0f + __expf(-x));
}

// pack two f32 -> one 4-byte LDS store (v_cvt_pk_bf16_f32 + ds_store_b32)
__device__ __forceinline__ void st_pk(__bf16* p, float a, float b) {
    *(v2bf*)p = (v2bf){(__bf16)a, (__bf16)b};
}

// A fragment (16x32 MxK, row-major tile): two ds_load_b128 (ISA 7.12.2).
__device__ __forceinline__ v16bf frag_a(const __bf16* tile, int rowBase, int lane) {
    const int m  = lane & 15;
    const int hi = lane >> 4;
    const __bf16* r = tile + (rowBase + m) * PITCH + hi * 8;
    const v8bf lo = *(const v8bf*)(r);
    const v8bf hh = *(const v8bf*)(r + 16);
    return __builtin_shufflevector(lo, hh, 0, 1, 2, 3, 4, 5, 6, 7,
                                           8, 9, 10, 11, 12, 13, 14, 15);
}

// B fragment from TRANSPOSED tile (rows = N, cols = K): two ds_load_b128.
__device__ __forceinline__ v16bf frag_bt(const __bf16* tileT, int rowBase, int lane) {
    const int n  = lane & 15;
    const int hi = lane >> 4;
    const __bf16* r = tileT + (rowBase + n) * PITCH + hi * 8;
    const v8bf lo = *(const v8bf*)(r);
    const v8bf hh = *(const v8bf*)(r + 16);
    return __builtin_shufflevector(lo, hh, 0, 1, 2, 3, 4, 5, 6, 7,
                                           8, 9, 10, 11, 12, 13, 14, 15);
}

// ===========================================================================
// Kernel 1: xz[b, 0:1024, l] = W_in(1024x256) @ x[b, :, l]
// 8 waves; block tile 128 rows x 64 cols; wave tile 16x64 (4 acc, frag_a
// reused across 4 WMMAs); double-buffered LDS, 1 barrier per K-step.
// ===========================================================================
__global__ __launch_bounds__(256)
void k_gemm_in(const float* __restrict__ x, const float* __restrict__ Win,
               float* __restrict__ xz) {
    __shared__ __align__(16) __bf16 sW[2][128 * PITCH];
    __shared__ __align__(16) __bf16 sXt[2][64 * PITCH];   // transposed: [l][k]

    const int tid   = threadIdx.x;
    const int wave  = tid >> 5;
    const int lane  = tid & 31;
    const int Lbase = blockIdx.x * 64;
    const int Mbase = blockIdx.y * 128;
    const int b     = blockIdx.z;

    auto stage = [&](int buf, int kk) {
#pragma unroll
        for (int it = 0; it < 8; ++it) {            // 128x32 W tile: 2048 pairs
            const int e = tid + it * 256;
            const int r = e >> 4, c = (e & 15) * 2;
            const float2 w = *(const float2*)&Win[(size_t)(Mbase + r) * DMODEL + kk + c];
            st_pk(&sW[buf][r * PITCH + c], w.x, w.y);
        }
#pragma unroll
        for (int it = 0; it < 4; ++it) {            // 32k x 64l transposed: 1024 pairs
            const int e = tid + it * 256;
            const int l = e & 63, k = (e >> 6) * 2;
            const size_t g = ((size_t)b * DMODEL + kk + k) * SEQLEN + Lbase + l;
            st_pk(&sXt[buf][l * PITCH + k], x[g], x[g + SEQLEN]);
        }
    };

    v8f acc[4] = {{}, {}, {}, {}};
    stage(0, 0);
    __syncthreads();
#pragma unroll 1
    for (int s = 0; s < DMODEL / 32; ++s) {
        const int cur = s & 1;
        if (s + 1 < DMODEL / 32) stage(cur ^ 1, (s + 1) * 32);
        const v16bf a = frag_a(&sW[cur][0], wave * 16, lane);
#pragma unroll
        for (int t = 0; t < 4; ++t) {
            const v16bf bt = frag_bt(&sXt[cur][0], t * 16, lane);
            acc[t] = __builtin_amdgcn_wmma_f32_16x16x32_bf16(
                false, a, false, bt, (short)0, acc[t], false, false);
        }
        __syncthreads();
    }
    const int n = lane & 15, hi = lane >> 4;
#pragma unroll
    for (int v = 0; v < 8; ++v) {
        const int row = Mbase + wave * 16 + v + 8 * hi;
        float* o = xz + ((size_t)b * (2 * DINNER) + row) * SEQLEN + Lbase + n;
        o[0] = acc[0][v]; o[16] = acc[1][v]; o[32] = acc[2][v]; o[48] = acc[3][v];
    }
}

// ===========================================================================
// Kernel 2: depthwise conv(3, pad 1) + bias + SiLU on the u-half of xz
// ===========================================================================
__global__ __launch_bounds__(256)
void k_conv_silu(const float* __restrict__ xz, const float* __restrict__ Wdw,
                 const float* __restrict__ bdw, float* __restrict__ u) {
    const size_t idx = (size_t)blockIdx.x * blockDim.x + threadIdx.x;
    const size_t total = (size_t)BATCH * DINNER * SEQLEN;
    if (idx >= total) return;
    const int l = (int)(idx % SEQLEN);
    const int d = (int)((idx / SEQLEN) % DINNER);
    const int b = (int)(idx / ((size_t)SEQLEN * DINNER));

    const float* row = xz + ((size_t)b * (2 * DINNER) + d) * SEQLEN;
    float acc = bdw[d];
#pragma unroll
    for (int t = 0; t < 3; ++t) {
        const int ll = l + t - 1;
        if (ll >= 0 && ll < SEQLEN) acc = fmaf(row[ll], Wdw[d * 3 + t], acc);
    }
    u[idx] = acc * fast_sigmoid(acc);
}

// ===========================================================================
// Kernel 3: x_dbl[b, 0:48, l] = W_x(48x512) @ u[b, :, l]
// 6 waves (3 M-tiles x 2 L-groups); wave tile 16x32 (2 acc); double-buffered.
// ===========================================================================
__global__ __launch_bounds__(192)
void k_gemm_xdbl(const float* __restrict__ u, const float* __restrict__ Wx,
                 float* __restrict__ xdbl) {
    __shared__ __align__(16) __bf16 sW[2][48 * PITCH];
    __shared__ __align__(16) __bf16 sUt[2][64 * PITCH];   // transposed: [l][k]

    const int tid   = threadIdx.x;
    const int wave  = tid >> 5;
    const int lane  = tid & 31;
    const int mi    = wave % 3;
    const int li    = wave / 3;          // 0..1, 32 cols each
    const int Lbase = blockIdx.x * 64;
    const int b     = blockIdx.y;

    auto stage = [&](int buf, int kk) {
#pragma unroll
        for (int it = 0; it < 4; ++it) {            // 48x32 W tile: 768 pairs
            const int e = tid + it * 192;
            const int r = e >> 4, c = (e & 15) * 2;
            const float2 w = *(const float2*)&Wx[(size_t)r * DINNER + kk + c];
            st_pk(&sW[buf][r * PITCH + c], w.x, w.y);
        }
#pragma unroll
        for (int it = 0; it < 6; ++it) {            // 32k x 64l transposed: 1024 pairs
            const int e = tid + it * 192;
            if (e < 1024) {
                const int l = e & 63, k = (e >> 6) * 2;
                const size_t g = ((size_t)b * DINNER + kk + k) * SEQLEN + Lbase + l;
                st_pk(&sUt[buf][l * PITCH + k], u[g], u[g + SEQLEN]);
            }
        }
    };

    v8f acc[2] = {{}, {}};
    stage(0, 0);
    __syncthreads();
#pragma unroll 1
    for (int s = 0; s < DINNER / 32; ++s) {
        const int cur = s & 1;
        if (s + 1 < DINNER / 32) stage(cur ^ 1, (s + 1) * 32);
        const v16bf a = frag_a(&sW[cur][0], mi * 16, lane);
#pragma unroll
        for (int t = 0; t < 2; ++t) {
            const v16bf bt = frag_bt(&sUt[cur][0], li * 32 + t * 16, lane);
            acc[t] = __builtin_amdgcn_wmma_f32_16x16x32_bf16(
                false, a, false, bt, (short)0, acc[t], false, false);
        }
        __syncthreads();
    }
    const int n = lane & 15, hi = lane >> 4;
#pragma unroll
    for (int v = 0; v < 8; ++v) {
        const int row = mi * 16 + v + 8 * hi;
        float* o = xdbl + ((size_t)b * XROWS + row) * SEQLEN + Lbase + li * 32 + n;
        o[0] = acc[0][v]; o[16] = acc[1][v];
    }
}

// ===========================================================================
// Kernel 4: delta = softplus(W_dt(512x16) @ x_dbl[:, 0:16, :] + b_dt)
// ===========================================================================
__global__ __launch_bounds__(256)
void k_delta(const float* __restrict__ xdbl, const float* __restrict__ Wdt,
             const float* __restrict__ bdt, float* __restrict__ delta) {
    const int l = blockIdx.x * 256 + threadIdx.x;
    const int d = blockIdx.y;
    const int b = blockIdx.z;

    float acc = bdt[d];
#pragma unroll
    for (int r = 0; r < DTRANK; ++r) {
        acc = fmaf(Wdt[d * DTRANK + r],
                   xdbl[((size_t)b * XROWS + r) * SEQLEN + l], acc);
    }
    const float sp = (acc > 20.0f) ? acc : __logf(1.0f + __expf(acc));
    delta[((size_t)b * DINNER + d) * SEQLEN + l] = sp;
}

// ===========================================================================
// Kernel 5: selective scan. One thread per (b, d); h[16] in registers.
// B/C tiles DMA'd with pipelined global_load_async_to_lds_b128: issue chunk
// c+1's copies, s_wait_asynccnt 2 -> chunk c complete (in-order), compute.
// ===========================================================================
#define CHUNK 64
__global__ __launch_bounds__(256)
void k_scan(const float* __restrict__ delta, const float* __restrict__ u,
            const float* __restrict__ xz, const float* __restrict__ xdbl,
            const float* __restrict__ Alog, const float* __restrict__ Dp,
            float* __restrict__ y) {
    __shared__ __align__(16) float sB[2][DSTATE][CHUNK];
    __shared__ __align__(16) float sC[2][DSTATE][CHUNK];

    const int tid = threadIdx.x;
    const int b   = blockIdx.x >> 1;
    const int d   = ((blockIdx.x & 1) << 8) + tid;

    float Av[DSTATE], h[DSTATE];
#pragma unroll
    for (int n = 0; n < DSTATE; ++n) {
        Av[n] = -__expf(Alog[d * DSTATE + n]);
        h[n]  = 0.0f;
    }
    const float Dd = Dp[d];

    const float* dRow  = delta + ((size_t)b * DINNER + d) * SEQLEN;
    const float* uRow  = u     + ((size_t)b * DINNER + d) * SEQLEN;
    const float* zRow  = xz    + ((size_t)b * (2 * DINNER) + DINNER + d) * SEQLEN;
    float*       yRow  = y     + ((size_t)b * DINNER + d) * SEQLEN;
    const float* Bbase = xdbl + ((size_t)b * XROWS + DTRANK) * SEQLEN;
    const float* Cbase = xdbl + ((size_t)b * XROWS + DTRANK + DSTATE) * SEQLEN;

    // per-thread slot of the tile copy: 4 consecutive floats (16 B per lane)
    const int sn = (tid * 4) >> 6;
    const int sc = (tid * 4) & (CHUNK - 1);
    const unsigned int ldsB[2] = {(unsigned int)(uintptr_t)&sB[0][sn][sc],
                                  (unsigned int)(uintptr_t)&sB[1][sn][sc]};
    const unsigned int ldsC[2] = {(unsigned int)(uintptr_t)&sC[0][sn][sc],
                                  (unsigned int)(uintptr_t)&sC[1][sn][sc]};

    auto issue = [&](int buf, int lc) {
        const unsigned long long gB =
            (unsigned long long)(uintptr_t)&Bbase[(size_t)sn * SEQLEN + lc + sc];
        const unsigned long long gC =
            (unsigned long long)(uintptr_t)&Cbase[(size_t)sn * SEQLEN + lc + sc];
        asm volatile("global_load_async_to_lds_b128 %0, %1, off"
                     :: "v"(ldsB[buf]), "v"(gB) : "memory");
        asm volatile("global_load_async_to_lds_b128 %0, %1, off"
                     :: "v"(ldsC[buf]), "v"(gC) : "memory");
    };

    issue(0, 0);
#pragma unroll 1
    for (int lc = 0, c = 0; lc < SEQLEN; lc += CHUNK, ++c) {
        const int cur = c & 1;
        if (lc + CHUNK < SEQLEN) {
            issue(cur ^ 1, lc + CHUNK);
            asm volatile("s_wait_asynccnt 0x2" ::: "memory");  // older 2 done
        } else {
            asm volatile("s_wait_asynccnt 0x0" ::: "memory");
        }
        __syncthreads();
        if (lc + CHUNK < SEQLEN) {             // gfx1250 global_prefetch_b8
            __builtin_prefetch(&dRow[lc + CHUNK], 0, 1);
            __builtin_prefetch(&uRow[lc + CHUNK], 0, 1);
            __builtin_prefetch(&zRow[lc + CHUNK], 0, 1);
        }

#pragma unroll 1
        for (int j4 = 0; j4 < CHUNK / 4; ++j4) {
            const int l0 = lc + j4 * 4;
            const float4 dv = *(const float4*)&dRow[l0];
            const float4 uv = *(const float4*)&uRow[l0];
            const float4 zv = *(const float4*)&zRow[l0];
            float dl[4], ul[4], zl[4], yl[4];
            *(float4*)dl = dv; *(float4*)ul = uv; *(float4*)zl = zv;
#pragma unroll
            for (int e = 0; e < 4; ++e) {
                const int li = j4 * 4 + e;
                const float dlt = dl[e], uu = ul[e];
                float accY = 0.0f;
#pragma unroll
                for (int n = 0; n < DSTATE; ++n) {
                    const float dA = __expf(dlt * Av[n]);
                    h[n] = fmaf(dA, h[n], dlt * sB[cur][n][li] * uu);
                    accY = fmaf(h[n], sC[cur][n][li], accY);
                }
                const float yv = fmaf(Dd, uu, accY);
                yl[e] = yv * (zl[e] * fast_sigmoid(zl[e]));
            }
            *(float4*)&yRow[l0] = *(float4*)yl;
        }
        __syncthreads();   // readers of buf cur done before it is re-issued
    }
}

// ===========================================================================
// Kernel 6: out[b, 0:256, l] = W_out(256x512) @ y[b, :, l] + x (residual)
// Same geometry as kernel 1; K = 512.
// ===========================================================================
__global__ __launch_bounds__(256)
void k_gemm_out(const float* __restrict__ y, const float* __restrict__ Wout,
                const float* __restrict__ x, float* __restrict__ out) {
    __shared__ __align__(16) __bf16 sW[2][128 * PITCH];
    __shared__ __align__(16) __bf16 sYt[2][64 * PITCH];   // transposed: [l][k]

    const int tid   = threadIdx.x;
    const int wave  = tid >> 5;
    const int lane  = tid & 31;
    const int Lbase = blockIdx.x * 64;
    const int Mbase = blockIdx.y * 128;
    const int b     = blockIdx.z;

    auto stage = [&](int buf, int kk) {
#pragma unroll
        for (int it = 0; it < 8; ++it) {
            const int e = tid + it * 256;
            const int r = e >> 4, c = (e & 15) * 2;
            const float2 w = *(const float2*)&Wout[(size_t)(Mbase + r) * DINNER + kk + c];
            st_pk(&sW[buf][r * PITCH + c], w.x, w.y);
        }
#pragma unroll
        for (int it = 0; it < 4; ++it) {
            const int e = tid + it * 256;
            const int l = e & 63, k = (e >> 6) * 2;
            const size_t g = ((size_t)b * DINNER + kk + k) * SEQLEN + Lbase + l;
            st_pk(&sYt[buf][l * PITCH + k], y[g], y[g + SEQLEN]);
        }
    };

    v8f acc[4] = {{}, {}, {}, {}};
    stage(0, 0);
    __syncthreads();
#pragma unroll 1
    for (int s = 0; s < DINNER / 32; ++s) {
        const int cur = s & 1;
        if (s + 1 < DINNER / 32) stage(cur ^ 1, (s + 1) * 32);
        const v16bf a = frag_a(&sW[cur][0], wave * 16, lane);
#pragma unroll
        for (int t = 0; t < 4; ++t) {
            const v16bf bt = frag_bt(&sYt[cur][0], t * 16, lane);
            acc[t] = __builtin_amdgcn_wmma_f32_16x16x32_bf16(
                false, a, false, bt, (short)0, acc[t], false, false);
        }
        __syncthreads();
    }
    const int n = lane & 15, hi = lane >> 4;
#pragma unroll
    for (int v = 0; v < 8; ++v) {
        const int row = Mbase + wave * 16 + v + 8 * hi;
        const size_t o = ((size_t)b * DMODEL + row) * SEQLEN + Lbase + n;
        out[o]      = acc[0][v] + x[o];
        out[o + 16] = acc[1][v] + x[o + 16];
        out[o + 32] = acc[2][v] + x[o + 32];
        out[o + 48] = acc[3][v] + x[o + 48];
    }
}

// ===========================================================================
extern "C" void kernel_launch(void* const* d_in, const int* in_sizes, int n_in,
                              void* d_out, int out_size, void* d_ws,
                              size_t ws_size, hipStream_t stream) {
    const float* x    = (const float*)d_in[0];
    const float* Win  = (const float*)d_in[1];
    const float* Wdw  = (const float*)d_in[2];
    const float* bdw  = (const float*)d_in[3];
    const float* Wx   = (const float*)d_in[4];
    const float* Wdt  = (const float*)d_in[5];
    const float* bdt  = (const float*)d_in[6];
    const float* Alog = (const float*)d_in[7];
    const float* Dp   = (const float*)d_in[8];
    const float* Wout = (const float*)d_in[9];
    float* out = (float*)d_out;
    float* ws  = (float*)d_ws;

    float* xz   = ws + OFF_XZ;
    float* u    = ws + OFF_U;
    float* xdbl = ws + OFF_XD;
    float* dlt  = ws + OFF_DL;
    float* yv   = ws + OFF_Y;

    k_gemm_in<<<dim3(SEQLEN / 64, (2 * DINNER) / 128, BATCH), 256, 0, stream>>>(
        x, Win, xz);
    {
        const size_t total = (size_t)BATCH * DINNER * SEQLEN;
        k_conv_silu<<<(unsigned)((total + 255) / 256), 256, 0, stream>>>(
            xz, Wdw, bdw, u);
    }
    k_gemm_xdbl<<<dim3(SEQLEN / 64, BATCH), 192, 0, stream>>>(u, Wx, xdbl);
    k_delta<<<dim3(SEQLEN / 256, DINNER, BATCH), 256, 0, stream>>>(
        xdbl, Wdt, bdt, dlt);
    k_scan<<<dim3(BATCH * (DINNER / 256)), 256, 0, stream>>>(
        dlt, u, xz, xdbl, Alog, Dp, yv);
    k_gemm_out<<<dim3(SEQLEN / 64, DMODEL / 128, BATCH), 256, 0, stream>>>(
        yv, Wout, x, out);
}
